// RNN_LSTM_attention_90907277787353
// MI455X (gfx1250) — compile-verified, matrix operand order checked by reading, not run on hardware
//
#include <hip/hip_runtime.h>
#include <hip/hip_bf16.h>

// ---------------------------------------------------------------------------
// Shapes: B=32, T=256, I=128, H=128, C=64
// ---------------------------------------------------------------------------
typedef _Float16 h16;
typedef __attribute__((ext_vector_type(8)))  _Float16 v8h;
typedef __attribute__((ext_vector_type(16))) _Float16 v16h;
typedef __attribute__((ext_vector_type(8)))  float    v8f;

#define WMMA_F32_F16(a, b, c) \
  __builtin_amdgcn_wmma_f32_16x16x32_f16(false, (a), false, (b), (short)0, (c), false, false)

// Async global->LDS 16-byte copy (CDNA5 VGLOBAL op 98, tracked by ASYNCcnt).
// lds_off: LDS byte address (low 32 bits of the generic pointer to __shared__,
// per ISA 10.2). gptr: per-lane global address.
__device__ __forceinline__ void async_ld_b128(unsigned int lds_off, const void* gptr) {
  asm volatile("global_load_async_to_lds_b128 %0, %1, off"
               :: "v"(lds_off), "v"(gptr)
               : "memory");
}
__device__ __forceinline__ void wait_async0() {
  asm volatile("s_wait_asynccnt 0" ::: "memory");
}
__device__ __forceinline__ unsigned int lds_off32(const void* p) {
  return (unsigned int)(unsigned long long)p;
}

// NT fragment loader: source is row-major [rows][ld] f16, fragment covers
// 16 rows starting at row0 and K-range [k0, k0+32).
// Lane layout (per ISA 16-bit A layout, used symmetrically for B):
//   row = row0 + (lane & 15); K(e) = (e&7) + 8*(lane>>4) + 16*(e>>3)
__device__ __forceinline__ v16h ld_frag(const h16* base, int ld, int row0, int k0) {
  int lane = threadIdx.x & 31;
  int r  = row0 + (lane & 15);
  int hi = lane >> 4;
  const h16* q = base + r * ld + k0 + hi * 8;
  v8h lo = *(const v8h*)(q);
  v8h hh = *(const v8h*)(q + 16);
  v16h v;
#pragma unroll
  for (int i = 0; i < 8; ++i) { v[i] = lo[i]; v[i + 8] = hh[i]; }
  return v;
}

__device__ __forceinline__ float fsigmoid(float x) {
  return 1.0f / (1.0f + __expf(-x));
}
__device__ __forceinline__ float ftanh(float x) {
  x = fminf(15.0f, fmaxf(-15.0f, x));
  float e = __expf(2.0f * x);
  return (e - 1.0f) / (e + 1.0f);
}

// ---------------------------------------------------------------------------
// Kernel 1: precision conversion / weight transposition / bias fusion
// ---------------------------------------------------------------------------
__global__ __launch_bounds__(256) void prep_kernel(
    const float* __restrict__ x, const float* __restrict__ Wih,
    const float* __restrict__ Whh, const float* __restrict__ bih,
    const float* __restrict__ bhh, const float* __restrict__ w1,
    const float* __restrict__ fcW,
    h16* __restrict__ xf, h16* __restrict__ wih16, h16* __restrict__ whh16,
    h16* __restrict__ w1t16, h16* __restrict__ fcw16, float* __restrict__ bsum) {
  int i = blockIdx.x * 256 + threadIdx.x;
  if (i < 8192 * 128) xf[i] = (h16)x[i];
  if (i < 512 * 128) { wih16[i] = (h16)Wih[i]; whh16[i] = (h16)Whh[i]; }
  if (i < 256 * 128) {
    // w_t_1 is (256,128); rows 0..127 = W1a, 128..255 = W1b. Store transposed
    // so h@W1{a,b} becomes an NT GEMM: w1t16[n][k] = w1[k + off][n&127]
    int n = i >> 7, k = i & 127;
    int off = (n >= 128) ? 128 : 0;
    w1t16[i] = (h16)w1[(k + off) * 128 + (n & 127)];
  }
  if (i < 64 * 128) fcw16[i] = (h16)fcW[i];
  if (i < 512) bsum[i] = bih[i] + bhh[i];
}

// ---------------------------------------------------------------------------
// Kernel 2 (Phase A): xW[m][n] = x[m]@W_ih^T[n] + (b_ih+b_hh)[n]
// M=8192 (b*256+t), N=512, K=128.  Grid (128, 4), 8 waves/WG.
// ---------------------------------------------------------------------------
__global__ __launch_bounds__(256) void gemm_xw(
    const h16* __restrict__ xf, const h16* __restrict__ wih16,
    const float* __restrict__ bsum, float* __restrict__ xW) {
  int lane = threadIdx.x & 31, w = threadIdx.x >> 5;
  int mt = w >> 1, ntb = (w & 1) * 4;
  int m0 = blockIdx.x * 64 + mt * 16;
  int nblk = blockIdx.y * 128;
  v8f acc[4] = {};
#pragma unroll
  for (int k0 = 0; k0 < 128; k0 += 32) {
    v16h a = ld_frag(xf, 128, m0, k0);
#pragma unroll
    for (int j = 0; j < 4; ++j) {
      v16h b = ld_frag(wih16, 128, nblk + (ntb + j) * 16, k0);
      acc[j] = WMMA_F32_F16(a, b, acc[j]);
    }
  }
  int hi = lane >> 4, nn = lane & 15;
#pragma unroll
  for (int j = 0; j < 4; ++j) {
    int n = nblk + (ntb + j) * 16 + nn;
    float bv = bsum[n];
#pragma unroll
    for (int r = 0; r < 8; ++r) {
      int m = m0 + r + 8 * hi;
      xW[m * 512 + n] = acc[j][r] + bv;
    }
  }
}

// ---------------------------------------------------------------------------
// Kernel 3 (Phase B): serial LSTM recurrence. One persistent workgroup,
// 8 waves, 256 dependent steps. Per step:
//   (async-stage xW[:,t,:] into LDS, overlapped with gemm1)
//   gates = h @ W_hh^T                  (WMMA M=32,N=512,K=128)
//   LSTM pointwise -> h,c               (VALU, exp-based sigmoid/tanh)
//   [hp|p] = h @ [W1a ; W1b]            (WMMA M=32,N=256,K=128)
// ---------------------------------------------------------------------------
__global__ __launch_bounds__(256) void lstm_seq(
    const float* __restrict__ xW, const h16* __restrict__ whh16,
    const h16* __restrict__ w1t16, h16* __restrict__ hT16,
    float* __restrict__ hall, float* __restrict__ hp, float* __restrict__ pp) {
  __shared__ alignas(16) float gates[32 * 512];  // 64 KB
  __shared__ alignas(16) float xw_l[32 * 512];   // 64 KB (async-staged xW slice)
  __shared__ alignas(16) h16   hsh[32 * 128];    //  8 KB
  __shared__ alignas(16) float cf[32 * 128];     // 16 KB
  int tid = threadIdx.x, lane = tid & 31, w = tid >> 5;
  int hi = lane >> 4, nn = lane & 15;

  for (int i = tid; i < 32 * 128; i += 256) { hsh[i] = (h16)0.0f; cf[i] = 0.0f; }
  __syncthreads();

  for (int t = 0; t < 256; ++t) {
    // ---- async-stage xW[:, t, :] (32x512 f32 = 64KB) under gemm1 ----
#pragma unroll
    for (int j = 0; j < 16; ++j) {           // 4096 16B-units / 256 threads
      int u = tid + j * 256;                 // unit = 4 floats
      int bb = u >> 7, cu = u & 127;         // 128 units per 512-float row
      async_ld_b128(lds_off32(&xw_l[u * 4]),
                    &xW[(size_t)((bb << 8) + t) * 512 + cu * 4]);
    }
    // ---- gemm1: h @ W_hh^T -> gates (LDS) ----
    {
      int ntb = w * 4;
      v8f acc[2][4] = {};
#pragma unroll
      for (int k0 = 0; k0 < 128; k0 += 32) {
        v16h a0 = ld_frag(hsh, 128, 0, k0);
        v16h a1 = ld_frag(hsh, 128, 16, k0);
#pragma unroll
        for (int j = 0; j < 4; ++j) {
          v16h b = ld_frag(whh16, 128, (ntb + j) * 16, k0);
          acc[0][j] = WMMA_F32_F16(a0, b, acc[0][j]);
          acc[1][j] = WMMA_F32_F16(a1, b, acc[1][j]);
        }
      }
#pragma unroll
      for (int mtj = 0; mtj < 2; ++mtj)
#pragma unroll
        for (int j = 0; j < 4; ++j) {
          int n = (ntb + j) * 16 + nn;
#pragma unroll
          for (int r = 0; r < 8; ++r) {
            int m = mtj * 16 + r + 8 * hi;
            gates[m * 512 + n] = acc[mtj][j][r];
          }
        }
    }
    wait_async0();
    __syncthreads();
    // ---- pointwise LSTM cell update ----
    for (int i = tid; i < 4096; i += 256) {
      int b = i >> 7, h = i & 127;
      const float* xr = xw_l + b * 512;
      float g_i = gates[b * 512 + h]       + xr[h];
      float g_f = gates[b * 512 + h + 128] + xr[h + 128];
      float g_g = gates[b * 512 + h + 256] + xr[h + 256];
      float g_o = gates[b * 512 + h + 384] + xr[h + 384];
      float cn = fsigmoid(g_f) * cf[i] + fsigmoid(g_i) * ftanh(g_g);
      float hn = fsigmoid(g_o) * ftanh(cn);
      cf[i] = cn;
      hsh[i] = (h16)hn;
      hT16[(b * 128 + h) * 256 + t] = (h16)hn;
      hall[(t * 32 + b) * 128 + h] = hn;
    }
    __syncthreads();
    // ---- gemm2: h @ [W1a ; W1b] -> hp, p (global) ----
    {
      int ntb = w * 2;
      v8f acc[2][2] = {};
#pragma unroll
      for (int k0 = 0; k0 < 128; k0 += 32) {
        v16h a0 = ld_frag(hsh, 128, 0, k0);
        v16h a1 = ld_frag(hsh, 128, 16, k0);
#pragma unroll
        for (int j = 0; j < 2; ++j) {
          v16h b = ld_frag(w1t16, 128, (ntb + j) * 16, k0);
          acc[0][j] = WMMA_F32_F16(a0, b, acc[0][j]);
          acc[1][j] = WMMA_F32_F16(a1, b, acc[1][j]);
        }
      }
#pragma unroll
      for (int mtj = 0; mtj < 2; ++mtj)
#pragma unroll
        for (int j = 0; j < 2; ++j) {
          int n = (ntb + j) * 16 + nn;
          float* dst = (n < 128) ? hp : pp;
          int nh = n & 127;
#pragma unroll
          for (int r = 0; r < 8; ++r) {
            int m = mtj * 16 + r + 8 * hi;  // batch index
            dst[(t * 32 + m) * 128 + nh] = acc[mtj][j][r];
          }
        }
    }
    // No trailing barrier needed: next-iter xw_l/gates writes are after the
    // barrier that followed the pointwise reads; hsh rewrites are after
    // next-iter gemm1's barrier.
  }
}

// ---------------------------------------------------------------------------
// Kernel 4 (Phase C): attention, fully parallel after phase B.
// Per (b, 64-row t-block): async-stage hp/p tiles into LDS, build masked score
// tile S[t][s] = (s<t) ? tanh(hp[t]+p[s]) . w2 : 0 (f16, LDS), then WMMA
// attn += S @ H (B-frags from hT16, s-contiguous). Fuses out_pre = 2h + attn.
// ---------------------------------------------------------------------------
__global__ __launch_bounds__(256) void attn_kernel(
    const float* __restrict__ hp, const float* __restrict__ pp,
    const float* __restrict__ hall, const h16* __restrict__ hT16,
    const float* __restrict__ w2, h16* __restrict__ pre16) {
  int b = blockIdx.y;
  int t0 = blockIdx.x * 64;
  __shared__ alignas(16) float hp_l[64 * 128];  // 32 KB
  __shared__ alignas(16) float p_l[32 * 128];   // 16 KB
  __shared__ alignas(16) h16   S_l[64 * 32];    //  4 KB
  __shared__ float w2_l[128];
  int tid = threadIdx.x, lane = tid & 31, w = tid >> 5;
  int mt = w >> 1, ntb = (w & 1) * 4;

  // async-stage hp block: 2048 16B-units, 8 per thread
#pragma unroll
  for (int j = 0; j < 8; ++j) {
    int u = tid + j * 256;
    int tl = u >> 5, cu = u & 31;            // 32 units per 128-float row
    async_ld_b128(lds_off32(&hp_l[u * 4]),
                  &hp[(size_t)((t0 + tl) * 32 + b) * 128 + cu * 4]);
  }
  if (tid < 128) w2_l[tid] = w2[tid];

  v8f acc[4] = {};
  for (int s0 = 0; s0 < 256; s0 += 32) {
    if (s0 > t0 + 62) break;  // uniform: whole chunk masked out
    __syncthreads();          // prior WMMA reads of S_l done before overwrite
    // async-stage p chunk: 1024 16B-units, 4 per thread
#pragma unroll
    for (int j = 0; j < 4; ++j) {
      int u = tid + j * 256;
      int sl = u >> 5, cu = u & 31;
      async_ld_b128(lds_off32(&p_l[u * 4]),
                    &pp[(size_t)((s0 + sl) * 32 + b) * 128 + cu * 4]);
    }
    wait_async0();            // covers hp_l on first chunk too
    __syncthreads();
    // score tile (the irreducible tanh work)
    for (int e = tid; e < 2048; e += 256) {
      int tl = e >> 5, sl = e & 31;
      float v = 0.0f;
      if (s0 + sl < t0 + tl) {
        float dot = 0.0f;
        const float* hr = hp_l + tl * 128;
        const float* pr = p_l + sl * 128;
#pragma unroll 4
        for (int h = 0; h < 128; ++h) dot += ftanh(hr[h] + pr[h]) * w2_l[h];
        v = dot;
      }
      S_l[tl * 32 + sl] = (h16)v;
    }
    __syncthreads();
    v16h a = ld_frag(S_l, 32, mt * 16, 0);
#pragma unroll
    for (int j = 0; j < 4; ++j) {
      v16h bf = ld_frag(hT16 + (size_t)b * 128 * 256, 256, (ntb + j) * 16, s0);
      acc[j] = WMMA_F32_F16(a, bf, acc[j]);
    }
  }
  int hi = lane >> 4, nn = lane & 15;
#pragma unroll
  for (int j = 0; j < 4; ++j) {
    int h = (ntb + j) * 16 + nn;
#pragma unroll
    for (int r = 0; r < 8; ++r) {
      int tl = mt * 16 + r + 8 * hi;
      int t = t0 + tl;
      float v = acc[j][r] + 2.0f * hall[(t * 32 + b) * 128 + h];
      pre16[(size_t)((b << 8) + t) * 128 + h] = (h16)v;
    }
  }
}

// ---------------------------------------------------------------------------
// Kernel 5 (Phase D): out = pre @ fc_W^T + fc_b.  M=8192, N=64, K=128.
// ---------------------------------------------------------------------------
__global__ __launch_bounds__(256) void gemm_fc(
    const h16* __restrict__ pre16, const h16* __restrict__ fcw16,
    const float* __restrict__ fcb, float* __restrict__ out) {
  int lane = threadIdx.x & 31, w = threadIdx.x >> 5;
  int mt = w >> 1, ntb = (w & 1) * 2;
  int m0 = blockIdx.x * 64 + mt * 16;
  v8f acc[2] = {};
#pragma unroll
  for (int k0 = 0; k0 < 128; k0 += 32) {
    v16h a = ld_frag(pre16, 128, m0, k0);
#pragma unroll
    for (int j = 0; j < 2; ++j) {
      v16h b = ld_frag(fcw16, 128, (ntb + j) * 16, k0);
      acc[j] = WMMA_F32_F16(a, b, acc[j]);
    }
  }
  int hi = lane >> 4, nn = lane & 15;
#pragma unroll
  for (int j = 0; j < 2; ++j) {
    int n = (ntb + j) * 16 + nn;
    float bv = fcb[n];
#pragma unroll
    for (int r = 0; r < 8; ++r) {
      int m = m0 + r + 8 * hi;
      out[m * 64 + n] = acc[j][r] + bv;
    }
  }
}

// ---------------------------------------------------------------------------
extern "C" void kernel_launch(void* const* d_in, const int* in_sizes, int n_in,
                              void* d_out, int out_size, void* d_ws, size_t ws_size,
                              hipStream_t stream) {
  const float* x    = (const float*)d_in[0];
  const float* Wih  = (const float*)d_in[1];
  const float* Whh  = (const float*)d_in[2];
  const float* bih  = (const float*)d_in[3];
  const float* bhh  = (const float*)d_in[4];
  const float* w1   = (const float*)d_in[5];
  const float* w2   = (const float*)d_in[6];
  const float* fcW  = (const float*)d_in[7];
  const float* fcb  = (const float*)d_in[8];

  char* ws = (char*)d_ws;
  size_t o = 0;
  auto take = [&](size_t bytes) { char* p = ws + o; o += (bytes + 255) & ~(size_t)255; return p; };
  h16*   xf    = (h16*)  take(8192 * 128 * 2);          // x in f16
  h16*   wih16 = (h16*)  take(512 * 128 * 2);
  h16*   whh16 = (h16*)  take(512 * 128 * 2);
  h16*   w1t16 = (h16*)  take(256 * 128 * 2);           // [W1a^T ; W1b^T]
  h16*   fcw16 = (h16*)  take(64 * 128 * 2);
  float* bsum  = (float*)take(512 * 4);
  float* xW    = (float*)take((size_t)8192 * 512 * 4);  // 16 MB
  h16*   hT16  = (h16*)  take((size_t)32 * 128 * 256 * 2);  // [b][h][t]
  float* hall  = (float*)take((size_t)256 * 32 * 128 * 4);  // [t][b][h]
  float* hp    = (float*)take((size_t)256 * 32 * 128 * 4);
  float* pp    = (float*)take((size_t)256 * 32 * 128 * 4);
  h16*   pre16 = (h16*)  take((size_t)8192 * 128 * 2);

  prep_kernel<<<4096, 256, 0, stream>>>(x, Wih, Whh, bih, bhh, w1, fcW,
                                        xf, wih16, whh16, w1t16, fcw16, bsum);
  gemm_xw<<<dim3(128, 4), 256, 0, stream>>>(xf, wih16, bsum, xW);
  lstm_seq<<<1, 256, 0, stream>>>(xW, whh16, w1t16, hT16, hall, hp, pp);
  attn_kernel<<<dim3(4, 32), 256, 0, stream>>>(hp, pp, hall, hT16, w2, pre16);
  gemm_fc<<<128, 256, 0, stream>>>(pre16, fcw16, fcb, (float*)d_out);
}
